// MambaBlock_37374805409895
// MI455X (gfx1250) — matrix operand order repaired and checked by reading, not compiled, not run on hardware
//
#include <hip/hip_runtime.h>
#include <stdint.h>

// ---------------------------------------------------------------------------
// Mamba block forward for MI455X (gfx1250, wave32, WMMA).
// Pipeline: LN -> in_proj (WMMA bf16) -> conv+SiLU -> x_proj (WMMA) ->
//           dt_proj (WMMA) -> selective scan (wave32, 16 lanes/channel) ->
//           gate -> out_proj (WMMA) + residual.
// GEMM uses double-buffered LDS staged with GLOBAL_LOAD_ASYNC_TO_LDS_B128
// (ASYNCcnt protocol) when the builtin is available.
// ---------------------------------------------------------------------------

#define DM   1024
#define DI   2048
#define DSN  16          // d_state
#define DCV  4           // d_conv
#define DR   64          // dt_rank
#define BATCH 2
#define SEQ  1024
#define RROWS (BATCH*SEQ) // 2048
#define XPN  (DR + 2*DSN) // 96
#define EPSF 1e-5f

typedef __attribute__((ext_vector_type(16))) __bf16 v16bf;
typedef __attribute__((ext_vector_type(8)))  float  v8f;
typedef int v4i __attribute__((vector_size(16)));

#if __has_builtin(__builtin_amdgcn_global_load_async_to_lds_b128)
#define HAVE_ASYNC_LDS 1
#else
#define HAVE_ASYNC_LDS 0
#endif

static __device__ __forceinline__ uint16_t f2bf(float f) {
  // round-to-nearest-even f32 -> bf16 (bit trick)
  uint32_t u = __float_as_uint(f);
  uint32_t r = (u + 0x7FFFu + ((u >> 16) & 1u)) >> 16;
  return (uint16_t)r;
}

static __device__ __forceinline__ void async_copy_b128(const uint16_t* gp,
                                                       uint16_t* lp) {
#if HAVE_ASYNC_LDS
  __builtin_amdgcn_global_load_async_to_lds_b128(
      (__attribute__((address_space(1))) v4i*)gp,
      (__attribute__((address_space(3))) v4i*)lp, 0, 0);
#else
  *(uint4*)lp = *(const uint4*)gp;
#endif
}

static __device__ __forceinline__ void wait_stage() {
#if HAVE_ASYNC_LDS
#if __has_builtin(__builtin_amdgcn_s_wait_asynccnt)
  __builtin_amdgcn_s_wait_asynccnt(0);
#else
  asm volatile("s_wait_asynccnt 0x0" ::: "memory");
#endif
#endif
}

// ------------------------- small elementwise kernels ------------------------

__global__ void k_f32_to_bf16(const float* __restrict__ src,
                              uint16_t* __restrict__ dst, int n) {
  int i = blockIdx.x * blockDim.x + threadIdx.x;
  if (i < n) dst[i] = f2bf(src[i]);
}

__global__ void k_neg_exp(const float* __restrict__ src,
                          float* __restrict__ dst, int n) {
  int i = blockIdx.x * blockDim.x + threadIdx.x;
  if (i < n) dst[i] = -__expf(src[i]);
}

// ------------------------------- LayerNorm ---------------------------------
__global__ __launch_bounds__(256) void k_layernorm(
    const float* __restrict__ x, const float* __restrict__ w,
    const float* __restrict__ b, uint16_t* __restrict__ out) {
  int row = blockIdx.x;
  const float* xr = x + (size_t)row * DM;
  float v[4], s0 = 0.f, s1 = 0.f;
#pragma unroll
  for (int j = 0; j < 4; j++) {
    v[j] = xr[threadIdx.x + j * 256];
    s0 += v[j];
    s1 += v[j] * v[j];
  }
#pragma unroll
  for (int off = 16; off; off >>= 1) {
    s0 += __shfl_xor(s0, off, 32);
    s1 += __shfl_xor(s1, off, 32);
  }
  __shared__ float rs0[8], rs1[8];
  int wv = threadIdx.x >> 5, ln = threadIdx.x & 31;
  if (ln == 0) { rs0[wv] = s0; rs1[wv] = s1; }
  __syncthreads();
  s0 = 0.f; s1 = 0.f;
#pragma unroll
  for (int i = 0; i < 8; i++) { s0 += rs0[i]; s1 += rs1[i]; }
  float mu  = s0 * (1.0f / DM);
  float var = s1 * (1.0f / DM) - mu * mu;
  float rs  = rsqrtf(var + EPSF);
#pragma unroll
  for (int j = 0; j < 4; j++) {
    int c = threadIdx.x + j * 256;
    out[(size_t)row * DM + c] = f2bf((v[j] - mu) * rs * w[c] + b[c]);
  }
}

// ------------------------- bf16 WMMA tiled GEMM ----------------------------
// C[M,N] (f32) = A[M,K](bf16) @ W[N,K](bf16)^T  (+ optional f32 Cadd)
// 256 threads = 8 waves (4 M-strips x 2 N-halves); wave tile 32x64 ->
// 8 x v_wmma_f32_16x16x32_bf16 per K-step. BM=BN=128, BK=32, double-buffered
// LDS fed by async global->LDS copies. Fragment layouts used here (contiguous k, not ISA table order):
//   A 16x32 bf16: lane m=lane&15, g=lane>>4; halves0-15 = K g*16..g*16+15
//                 -> two ds_read_b128
//   B 32x16 bf16: lane n=lane&15, g=lane>>4; halves0-15 = K g*16..g*16+15
#define BM 128
#define BN 128
#define BK 32
#define LDP 40   // padded row stride in bf16 elems (bank-conflict relief)

static __device__ __forceinline__ void stage_tiles(
    const uint16_t* __restrict__ A, const uint16_t* __restrict__ W,
    uint16_t* sAb, uint16_t* sBb, int rowBase, int colBase,
    int N, int K, int k0, int tid) {
  // A tile: 128 rows x 32 bf16 = 8KB -> 2 x 16B chunks per thread
#pragma unroll
  for (int i = 0; i < 2; i++) {
    int chunk = tid * 2 + i;                 // 0..511
    int r = chunk >> 2, c8 = (chunk & 3) * 8;
    async_copy_b128(A + (size_t)(rowBase + r) * K + k0 + c8,
                    &sAb[r * LDP + c8]);
  }
  // B tile: 128 rows x 32 bf16; rows >= N only feed never-stored C columns,
  // so clamp the row (keeps addresses in-bounds, avoids divergence).
#pragma unroll
  for (int i = 0; i < 2; i++) {
    int chunk = tid * 2 + i;
    int r = chunk >> 2, c8 = (chunk & 3) * 8;
    int gn = colBase + r;
    gn = (gn < N) ? gn : (N - 1);
    async_copy_b128(W + (size_t)gn * K + k0 + c8, &sBb[r * LDP + c8]);
  }
}

__global__ __launch_bounds__(256) void k_gemm_bf16(
    const uint16_t* __restrict__ A,   // M x K  (bf16 bits)
    const uint16_t* __restrict__ W,   // N x K  (bf16 bits)
    const float* __restrict__ Cadd,   // optional M x N residual (may be null)
    float* __restrict__ Cout,         // M x N
    int M, int N, int K) {
  __shared__ uint16_t sA[2][BM * LDP];
  __shared__ uint16_t sB[2][BN * LDP];

  const int tid  = threadIdx.x;
  const int lane = tid & 31;
  const int wave = tid >> 5;
  const int g  = lane >> 4;     // lane group (0,1)
  const int nn = lane & 15;     // n (B/C/D) or m (A) within 16x16 tile
  const int wm = wave >> 1;     // 0..3  (32-row M strip)
  const int wn = wave & 1;      // 0..1  (64-col N half)
  const int rowBase = blockIdx.y * BM;
  const int colBase = blockIdx.x * BN;

  v8f acc[2][4] = {};           // 8 x (16x16 f32) per wave

  stage_tiles(A, W, sA[0], sB[0], rowBase, colBase, N, K, 0, tid);
  wait_stage();
  __syncthreads();

  int cur = 0;
  for (int k0 = 0; k0 < K; k0 += BK) {
    const int nxt = cur ^ 1;
    if (k0 + BK < K)
      stage_tiles(A, W, sA[nxt], sB[nxt], rowBase, colBase, N, K, k0 + BK, tid);

    // ---- batched fragment loads (one DS clause), then 8 back-to-back WMMAs
    v16bf af[2];
#pragma unroll
    for (int mi = 0; mi < 2; mi++) {
      const uint16_t* p = &sA[cur][(wm * 32 + mi * 16 + nn) * LDP + g * 16];
      __builtin_memcpy(&af[mi], p, 32);                  // K = g*16 .. +15
    }
    v16bf bfr[4];
#pragma unroll
    for (int t = 0; t < 4; t++) {
      const uint16_t* p = &sB[cur][(wn * 64 + t * 16 + nn) * LDP + g * 16];
      __builtin_memcpy(&bfr[t], p, 32);                  // K = g*16 .. +15
    }
#pragma unroll
    for (int mi = 0; mi < 2; mi++)
#pragma unroll
      for (int t = 0; t < 4; t++)
        acc[mi][t] = __builtin_amdgcn_wmma_f32_16x16x32_bf16(
            false, af[mi], false, bfr[t], (short)0, acc[mi][t], false, false);

    wait_stage();       // our async stores to buf[nxt] have landed
    __syncthreads();    // everyone: nxt staged, cur fully consumed
    cur = nxt;
  }

  // ---- store: D element (v,g,nn) -> row = v + g*8, col = nn within 16x16
#pragma unroll
  for (int mi = 0; mi < 2; mi++) {
#pragma unroll
    for (int t = 0; t < 4; t++) {
      int gc = colBase + wn * 64 + t * 16 + nn;
      if (gc < N) {
#pragma unroll
        for (int v = 0; v < 8; v++) {
          int gr = rowBase + wm * 32 + mi * 16 + g * 8 + v;
          size_t idx = (size_t)gr * N + gc;
          float val = acc[mi][t][v];
          if (Cadd) val += Cadd[idx];
          Cout[idx] = val;
        }
      }
    }
  }
}

// ------------------- causal depthwise conv (DC=4) + SiLU -------------------
__global__ void k_conv_silu(const float* __restrict__ xz,
                            const float* __restrict__ cw,
                            const float* __restrict__ cb,
                            float* __restrict__ u,
                            uint16_t* __restrict__ xb) {
  int i = blockIdx.x * blockDim.x + threadIdx.x;   // over RROWS*DI
  if (i >= RROWS * DI) return;
  int d = i & (DI - 1);
  int r = i >> 11;                                  // / DI
  int b = r / SEQ, l = r - b * SEQ;
  float acc = cb[d];
#pragma unroll
  for (int j = 0; j < DCV; j++) {
    int ls = l - (DCV - 1) + j;
    if (ls >= 0)
      acc += cw[d * DCV + j] * xz[(size_t)(b * SEQ + ls) * (2 * DI) + d];
  }
  float s = acc / (1.f + __expf(-acc));
  u[i]  = s;
  xb[i] = f2bf(s);
}

// ------------- extract dt columns (first DR of x_dbl) as bf16 --------------
__global__ void k_dt_extract(const float* __restrict__ xdbl,
                             uint16_t* __restrict__ dtb) {
  int i = blockIdx.x * blockDim.x + threadIdx.x;   // over RROWS*DR
  if (i >= RROWS * DR) return;
  int r = i >> 6, j = i & (DR - 1);
  dtb[i] = f2bf(xdbl[(size_t)r * XPN + j]);
}

// --------------------------- selective scan --------------------------------
// 2 channels per wave32; 16 lanes per channel (lane = state index s).
__global__ __launch_bounds__(256) void k_scan(
    const float* __restrict__ dt_raw,   // R x DI (pre-bias, pre-softplus)
    const float* __restrict__ dt_bias,  // DI
    const float* __restrict__ u,        // R x DI
    const float* __restrict__ xdbl,     // R x 96 (B at +64, C at +80)
    const float* __restrict__ Aneg,     // DI x 16  (= -exp(A_log))
    const float* __restrict__ Dp,       // DI
    const float* __restrict__ xz,       // R x 4096 (z at +DI)
    uint16_t* __restrict__ yg)          // R x DI  (gated output, bf16)
{
  const int lane = threadIdx.x & 31;
  const int wave = threadIdx.x >> 5;
  const int sub  = lane >> 4;           // which of 2 channels in wave
  const int s    = lane & 15;           // state index
  const int c    = blockIdx.x * 16 + wave * 2 + sub;   // 0..8191
  const int b    = c >> 11;             // / DI
  const int d    = c & (DI - 1);

  const float Ar  = Aneg[d * DSN + s];
  const float Dd  = Dp[d];
  const float dtb = dt_bias[d];
  float h = 0.f;

  for (int l = 0; l < SEQ; l++) {
    const int r = b * SEQ + l;
    float dtv = dt_raw[(size_t)r * DI + d] + dtb;
    dtv = (dtv > 20.f) ? dtv : log1pf(__expf(dtv));   // softplus
    const float uv = u[(size_t)r * DI + d];
    const float Bv = xdbl[(size_t)r * XPN + DR + s];
    const float Cv = xdbl[(size_t)r * XPN + DR + DSN + s];
    h = __expf(dtv * Ar) * h + (dtv * uv) * Bv;
    float y = h * Cv;
    y += __shfl_xor(y, 8, 32);
    y += __shfl_xor(y, 4, 32);
    y += __shfl_xor(y, 2, 32);
    y += __shfl_xor(y, 1, 32);
    if (s == 0) {
      const float z = xz[(size_t)r * (2 * DI) + DI + d];
      const float gate = z / (1.f + __expf(-z));
      yg[(size_t)r * DI + d] = f2bf((y + Dd * uv) * gate);
    }
  }
}

// ------------------------------- launcher ----------------------------------

extern "C" void kernel_launch(void* const* d_in, const int* in_sizes, int n_in,
                              void* d_out, int out_size, void* d_ws, size_t ws_size,
                              hipStream_t stream) {
  const float* hid  = (const float*)d_in[0];   // (B,L,DM)
  const float* nw   = (const float*)d_in[1];
  const float* nb   = (const float*)d_in[2];
  const float* inw  = (const float*)d_in[3];   // (2DI, DM)
  const float* cw   = (const float*)d_in[4];   // (DI,1,DC)
  const float* cb   = (const float*)d_in[5];
  const float* xpw  = (const float*)d_in[6];   // (96, DI)
  const float* dtw  = (const float*)d_in[7];   // (DI, DR)
  const float* dtbv = (const float*)d_in[8];   // (DI,)
  const float* alog = (const float*)d_in[9];   // (DI, 16)
  const float* Dp   = (const float*)d_in[10];  // (DI,)
  const float* outw = (const float*)d_in[11];  // (DM, DI)
  float* out = (float*)d_out;

  char* p = (char*)d_ws;
  auto alloc = [&](size_t bytes) -> char* {
    char* q = p;
    p += (bytes + 255) & ~(size_t)255;
    return q;
  };
  uint16_t* h_bf  = (uint16_t*)alloc((size_t)RROWS * DM * 2);
  uint16_t* w_in  = (uint16_t*)alloc((size_t)2 * DI * DM * 2);
  uint16_t* w_xp  = (uint16_t*)alloc((size_t)XPN * DI * 2);
  uint16_t* w_dt  = (uint16_t*)alloc((size_t)DI * DR * 2);
  uint16_t* w_out = (uint16_t*)alloc((size_t)DM * DI * 2);
  float*    Aneg  = (float*)alloc((size_t)DI * DSN * 4);
  float*    xzbuf = (float*)alloc((size_t)RROWS * 2 * DI * 4);
  float*    ubuf  = (float*)alloc((size_t)RROWS * DI * 4);
  uint16_t* x_bf  = (uint16_t*)alloc((size_t)RROWS * DI * 2);
  float*    xdbl  = (float*)alloc((size_t)RROWS * XPN * 4);
  uint16_t* dt_bf = (uint16_t*)alloc((size_t)RROWS * DR * 2);
  float*    dtraw = (float*)alloc((size_t)RROWS * DI * 4);
  uint16_t* yg_bf = (uint16_t*)alloc((size_t)RROWS * DI * 2);

  auto cvt = [&](const float* s, uint16_t* d2, int n) {
    k_f32_to_bf16<<<(n + 255) / 256, 256, 0, stream>>>(s, d2, n);
  };
  cvt(inw,  w_in,  2 * DI * DM);
  cvt(xpw,  w_xp,  XPN * DI);
  cvt(dtw,  w_dt,  DI * DR);
  cvt(outw, w_out, DM * DI);
  k_neg_exp<<<(DI * DSN + 255) / 256, 256, 0, stream>>>(alog, Aneg, DI * DSN);

  // 1) LayerNorm -> bf16
  k_layernorm<<<RROWS, 256, 0, stream>>>(hid, nw, nb, h_bf);

  // 2) in_proj: xz[2048,4096] = h @ in_proj_w^T   (K=1024)
  {
    dim3 g((2 * DI + BN - 1) / BN, (RROWS + BM - 1) / BM);
    k_gemm_bf16<<<g, 256, 0, stream>>>(h_bf, w_in, nullptr, xzbuf,
                                       RROWS, 2 * DI, DM);
  }

  // 3) causal depthwise conv + SiLU -> u(f32), x(bf16)
  k_conv_silu<<<(RROWS * DI + 255) / 256, 256, 0, stream>>>(xzbuf, cw, cb,
                                                            ubuf, x_bf);

  // 4) x_proj: x_dbl[2048,96] = x @ x_proj_w^T    (K=2048)
  {
    dim3 g((XPN + BN - 1) / BN, (RROWS + BM - 1) / BM);
    k_gemm_bf16<<<g, 256, 0, stream>>>(x_bf, w_xp, nullptr, xdbl,
                                       RROWS, XPN, DI);
  }

  // 5) dt columns -> bf16; dt_proj: dtraw[2048,2048] = dt @ dt_proj_w^T (K=64)
  k_dt_extract<<<(RROWS * DR + 255) / 256, 256, 0, stream>>>(xdbl, dt_bf);
  {
    dim3 g((DI + BN - 1) / BN, (RROWS + BM - 1) / BM);
    k_gemm_bf16<<<g, 256, 0, stream>>>(dt_bf, w_dt, nullptr, dtraw,
                                       RROWS, DI, DR);
  }

  // 6) selective scan + D-skip + SiLU(z) gate -> yg (bf16)
  k_scan<<<(2 * DI * BATCH) / 16, 256, 0, stream>>>(dtraw, dtbv, ubuf, xdbl,
                                                    Aneg, Dp, xzbuf, yg_bf);

  // 7) out_proj + residual -> d_out (f32)
  {
    dim3 g((DM + BN - 1) / BN, (RROWS + BM - 1) / BM);
    k_gemm_bf16<<<g, 256, 0, stream>>>(yg_bf, w_out, hid, out,
                                       RROWS, DM, DI);
  }
}
